// GraphNet_55499567399389
// MI455X (gfx1250) — matrix operand order repaired and checked by reading, not compiled, not run on hardware
//
#include <hip/hip_runtime.h>
#include <hip/hip_bf16.h>

// GraphNet (MetaLayer GNN) for MI455X / gfx1250.
// All GEMMs run as v_wmma_f32_16x16x32_bf16; activations staged in LDS as bf16,
// accumulation in f32. Each wave processes 2 M-tiles (32 rows) so every weight
// B-fragment load feeds two WMMAs (halves L2 weight traffic, doubles WMMA ILP).
// Scatter-means via f32 atomics; per-graph sums use 64-way partial buffers to
// kill contention on the B=16 target rows.

#define NN 50000
#define EE 800000
#define BB 16
#define DD 64
#define HH 128
#define LL 2
#define PP 64   // partial-buffer fan-out for per-graph reductions

typedef __attribute__((ext_vector_type(16))) __bf16 v16bf;
typedef __attribute__((ext_vector_type(8)))  float  v8f;

// ---------------- WMMA fragment helpers (layouts per CDNA5 ISA) ---

// A 16x32 bf16: lane m=lane&15; half=lane>>4: K chunks [half*8, +7] and [half*8+16, +7]
static __device__ __forceinline__ v16bf load_a_frag(const __bf16* A, int stride,
                                                    int row, int kofs, int half) {
  union { uint4 u[2]; v16bf v; } un;
  const __bf16* p = A + row * stride + kofs + half * 8;
  un.u[0] = *(const uint4*)p;
  un.u[1] = *(const uint4*)(p + 16);
  return un.v;
}

// B 32x16 bf16 from transposed weights Wt[N][K]: lane n=lane&15; half=lane>>4:
// K chunks [kofs + half*8, +7] and [kofs + half*8 + 16, +7]
static __device__ __forceinline__ v16bf load_b_frag(const __bf16* Wt, int KT,
                                                    int n, int kofs, int half) {
  union { uint4 u[2]; v16bf v; } un;
  const __bf16* p = Wt + (size_t)n * KT + kofs + half * 8;
  un.u[0] = *(const uint4*)p;
  un.u[1] = *(const uint4*)(p + 16);
  return un.v;
}

// One wave computes an (MT*16) x NT GEMM (K=KT) from an LDS bf16 tile,
// bias + ReLU, result staged back to LDS bf16 (row-major, stride os).
// B fragments are loaded once per (nt,ks) and reused for all MT row-tiles.
template <int KT, int NT, int MT>
static __device__ __forceinline__ void wgemm_relu_lds(
    const __bf16* A, int as,
    const __bf16* Wt, const float* bias,
    __bf16* O, int os, int lane) {
  constexpr int KS = KT / 32;
  constexpr int NTL = NT / 16;
  const int m = lane & 15, half = lane >> 4;
  v16bf af[MT][KS];
#pragma unroll
  for (int mt = 0; mt < MT; ++mt)
#pragma unroll
    for (int ks = 0; ks < KS; ++ks)
      af[mt][ks] = load_a_frag(A, as, mt * 16 + m, ks * 32, half);
#pragma unroll
  for (int nt = 0; nt < NTL; ++nt) {
    float bv = bias[nt * 16 + m];
    v8f acc[MT];
#pragma unroll
    for (int mt = 0; mt < MT; ++mt)
#pragma unroll
      for (int r = 0; r < 8; ++r) acc[mt][r] = bv;
#pragma unroll
    for (int ks = 0; ks < KS; ++ks) {
      v16bf bf = load_b_frag(Wt, KT, nt * 16 + m, ks * 32, half);
#pragma unroll
      for (int mt = 0; mt < MT; ++mt)
        acc[mt] = __builtin_amdgcn_wmma_f32_16x16x32_bf16(
            false, af[mt][ks], false, bf, (short)0, acc[mt], false, false);
    }
#pragma unroll
    for (int mt = 0; mt < MT; ++mt)
#pragma unroll
      for (int r = 0; r < 8; ++r) {
        float v = acc[mt][r] > 0.f ? acc[mt][r] : 0.f;
        O[(mt * 16 + half * 8 + r) * os + nt * 16 + m] = (__bf16)v;
      }
  }
}

// Same, but final f32 results handed to a functor f(nCol, mRow, value) (no ReLU).
template <int KT, int NT, int MT, class F>
static __device__ __forceinline__ void wgemm_apply(
    const __bf16* A, int as,
    const __bf16* Wt, const float* bias, int lane, F&& f) {
  constexpr int KS = KT / 32;
  constexpr int NTL = NT / 16;
  const int m = lane & 15, half = lane >> 4;
  v16bf af[MT][KS];
#pragma unroll
  for (int mt = 0; mt < MT; ++mt)
#pragma unroll
    for (int ks = 0; ks < KS; ++ks)
      af[mt][ks] = load_a_frag(A, as, mt * 16 + m, ks * 32, half);
#pragma unroll
  for (int nt = 0; nt < NTL; ++nt) {
    float bv = bias[nt * 16 + m];
    v8f acc[MT];
#pragma unroll
    for (int mt = 0; mt < MT; ++mt)
#pragma unroll
      for (int r = 0; r < 8; ++r) acc[mt][r] = bv;
#pragma unroll
    for (int ks = 0; ks < KS; ++ks) {
      v16bf bf = load_b_frag(Wt, KT, nt * 16 + m, ks * 32, half);
#pragma unroll
      for (int mt = 0; mt < MT; ++mt)
        acc[mt] = __builtin_amdgcn_wmma_f32_16x16x32_bf16(
            false, af[mt][ks], false, bf, (short)0, acc[mt], false, false);
    }
#pragma unroll
    for (int mt = 0; mt < MT; ++mt)
#pragma unroll
      for (int r = 0; r < 8; ++r)
        f(nt * 16 + m, mt * 16 + half * 8 + r, acc[mt][r]);
  }
}

// ---------------- Edge kernel: edge MLP (3) + node-MLP1 (2), fully fused ------
// Block = 32 threads (1 wave), 32 edges/block (2 M-tiles per wave).
__global__ void __launch_bounds__(32) gn_edge_kernel(
    const float* __restrict__ xin, const float* __restrict__ ein,
    const float* __restrict__ uin,
    const int* __restrict__ row, const int* __restrict__ col,
    const int* __restrict__ batch,
    const __bf16* __restrict__ w1, const float* __restrict__ b1,
    const __bf16* __restrict__ w2, const float* __restrict__ b2,
    const __bf16* __restrict__ w3, const float* __restrict__ b3,
    const __bf16* __restrict__ nw1, const float* __restrict__ nb1,
    const __bf16* __restrict__ nw2, const float* __restrict__ nb2,
    float* __restrict__ eout, float* __restrict__ agg_sum,
    float* __restrict__ e_agg_part) {
  __shared__ __bf16 IN[32 * 264];   // [32 edges][256] concat input, pad 8
  __shared__ __bf16 HA[32 * 136];   // stage A
  __shared__ __bf16 HB[32 * 136];   // stage B
  __shared__ int mrow[32], mcol[32], mb[32];

  const int lane = threadIdx.x;
  const int base = blockIdx.x * 32;
  float* eagg = e_agg_part + (size_t)(blockIdx.x & (PP - 1)) * BB * DD;

  {
    int e = base + lane; if (e >= EE) e = EE - 1;
    int r = row[e];
    mrow[lane] = r; mcol[lane] = col[e]; mb[lane] = batch[r];
  }
  __syncthreads();

  // Stage concat [x[row] | x[col] | edge_attr | u[eb]] as bf16.
  for (int q = lane; q < 32 * 64; q += 32) {
    int el = q >> 6, c4 = q & 63, seg = c4 >> 4, qi = c4 & 15;
    const float* src;
    if (seg == 0)      src = xin + (size_t)mrow[el] * DD + qi * 4;
    else if (seg == 1) src = xin + (size_t)mcol[el] * DD + qi * 4;
    else if (seg == 2) src = ein + ((size_t)base + el) * DD + qi * 4;
    else               src = uin + (size_t)mb[el] * DD + qi * 4;
    float4 fv = *(const float4*)src;
    union { __bf16 h[4]; uint2 u; } pk;
    pk.h[0] = (__bf16)fv.x; pk.h[1] = (__bf16)fv.y;
    pk.h[2] = (__bf16)fv.z; pk.h[3] = (__bf16)fv.w;
    *(uint2*)&IN[el * 264 + c4 * 4] = pk.u;
  }
  __syncthreads();

  // Edge MLP
  wgemm_relu_lds<256, 128, 2>(IN, 264, w1, b1, HA, 136, lane);
  __syncthreads();
  wgemm_relu_lds<128, 128, 2>(HA, 136, w2, b2, HB, 136, lane);
  __syncthreads();
  // Final edge layer: write e_new, e_agg partial atomics, stage into HA[:,64:128]
  wgemm_apply<128, 64, 2>(HB, 136, w3, b3, lane,
      [&](int n, int mr, float v) {
        size_t e = (size_t)base + mr;
        if (e < (size_t)EE) {
          eout[e * DD + n] = v;
          atomicAdd(&eagg[mb[mr] * DD + n], v);
        }
        HA[mr * 136 + 64 + n] = (__bf16)v;
      });
  // Copy x[row] into HA[:,0:64] -> node-MLP1 input = [x[row] | e_new]
  for (int i = lane; i < 32 * 64; i += 32) {
    int mr = i >> 6, c = i & 63;
    HA[mr * 136 + c] = IN[mr * 264 + c];
  }
  __syncthreads();

  // Node-MLP1
  wgemm_relu_lds<128, 128, 2>(HA, 136, nw1, nb1, HB, 136, lane);
  __syncthreads();
  wgemm_apply<128, 128, 2>(HB, 136, nw2, nb2, lane,
      [&](int n, int mr, float v) {
        if (base + mr < EE)
          atomicAdd(&agg_sum[(size_t)mcol[mr] * HH + n], v);
      });
}

// ---------------- Node kernel: node-MLP2 -------------------------------------
__global__ void __launch_bounds__(32) gn_node_kernel(
    const float* __restrict__ xin, const float* __restrict__ uin,
    const float* __restrict__ agg_sum, const float* __restrict__ col_cnt,
    const int* __restrict__ batch,
    const __bf16* __restrict__ w1, const float* __restrict__ b1,
    const __bf16* __restrict__ w2, const float* __restrict__ b2,
    float* __restrict__ xout, float* __restrict__ n_agg_part) {
  __shared__ __bf16 IN[32 * 264];
  __shared__ __bf16 HB[32 * 136];
  __shared__ int mb[32];
  __shared__ float mcoef[32];

  const int lane = threadIdx.x;
  const int base = blockIdx.x * 32;
  float* nagg = n_agg_part + (size_t)(blockIdx.x & (PP - 1)) * BB * DD;

  {
    int i = base + lane;
    if (i < NN) {
      mb[lane] = batch[i];
      mcoef[lane] = 1.f / fmaxf(col_cnt[i], 1.f);
    } else { mb[lane] = 0; mcoef[lane] = 0.f; }
  }
  __syncthreads();

  // Stage [x | agg_sum/cnt | u[batch]] (K=256)
  for (int q = lane; q < 32 * 64; q += 32) {
    int el = q >> 6, c4 = q & 63;
    int i = base + el;
    float4 fv = {0.f, 0.f, 0.f, 0.f};
    if (i < NN) {
      if (c4 < 16) fv = *(const float4*)(xin + (size_t)i * DD + c4 * 4);
      else if (c4 < 48) {
        fv = *(const float4*)(agg_sum + (size_t)i * HH + (c4 - 16) * 4);
        float s = mcoef[el];
        fv.x *= s; fv.y *= s; fv.z *= s; fv.w *= s;
      } else fv = *(const float4*)(uin + (size_t)mb[el] * DD + (c4 - 48) * 4);
    }
    union { __bf16 h[4]; uint2 u; } pk;
    pk.h[0] = (__bf16)fv.x; pk.h[1] = (__bf16)fv.y;
    pk.h[2] = (__bf16)fv.z; pk.h[3] = (__bf16)fv.w;
    *(uint2*)&IN[el * 264 + c4 * 4] = pk.u;
  }
  __syncthreads();

  wgemm_relu_lds<256, 128, 2>(IN, 264, w1, b1, HB, 136, lane);
  __syncthreads();
  wgemm_apply<128, 64, 2>(HB, 136, w2, b2, lane,
      [&](int n, int mr, float v) {
        int i = base + mr;
        if (i < NN) {
          xout[(size_t)i * DD + n] = v;
          atomicAdd(&nagg[mb[mr] * DD + n], v);
        }
      });
}

// ---------------- Global kernel: B=16 rows, one wave -------------------------
__global__ void __launch_bounds__(32) gn_global_kernel(
    const float* __restrict__ uin, const float* __restrict__ n_agg,
    const float* __restrict__ e_agg,
    const float* __restrict__ nb_cnt, const float* __restrict__ eb_cnt,
    const __bf16* __restrict__ w1, const float* __restrict__ b1,
    const __bf16* __restrict__ w2, const float* __restrict__ b2,
    float* __restrict__ uout) {
  __shared__ __bf16 IN[16 * 200];   // K=192, pad to 200 (16B-aligned rows)
  __shared__ __bf16 HB[16 * 136];
  __shared__ float ncf[16], ecf[16];

  const int t = threadIdx.x;
  if (t < 16) {
    ncf[t] = 1.f / fmaxf(nb_cnt[t], 1.f);
    ecf[t] = 1.f / fmaxf(eb_cnt[t], 1.f);
  }
  __syncthreads();

  for (int q = t; q < 16 * 48; q += 32) {
    int g = q / 48, c4 = q % 48;
    float4 fv;
    if (c4 < 16) fv = *(const float4*)(uin + (size_t)g * DD + c4 * 4);
    else if (c4 < 32) {
      fv = *(const float4*)(n_agg + (size_t)g * DD + (c4 - 16) * 4);
      float s = ncf[g]; fv.x *= s; fv.y *= s; fv.z *= s; fv.w *= s;
    } else {
      fv = *(const float4*)(e_agg + (size_t)g * DD + (c4 - 32) * 4);
      float s = ecf[g]; fv.x *= s; fv.y *= s; fv.z *= s; fv.w *= s;
    }
    union { __bf16 h[4]; uint2 u; } pk;
    pk.h[0] = (__bf16)fv.x; pk.h[1] = (__bf16)fv.y;
    pk.h[2] = (__bf16)fv.z; pk.h[3] = (__bf16)fv.w;
    *(uint2*)&IN[g * 200 + c4 * 4] = pk.u;
  }
  __syncthreads();

  wgemm_relu_lds<192, 128, 1>(IN, 200, w1, b1, HB, 136, t);
  __syncthreads();
  wgemm_apply<128, 64, 1>(HB, 136, w2, b2, t,
      [&](int n, int mr, float v) {
        if (mr < BB) uout[(size_t)mr * DD + n] = v;
      });
}

// ---------------- Utility kernels -------------------------------------------
__global__ void gn_convT(const float* __restrict__ src, __bf16* __restrict__ dst,
                         int K, int Nc) {
  int idx = blockIdx.x * blockDim.x + threadIdx.x;
  if (idx < K * Nc) {
    int k = idx / Nc, n = idx % Nc;
    dst[(size_t)n * K + k] = (__bf16)src[idx];   // transpose: Wt[N][K]
  }
}

__global__ void gn_counts(const int* __restrict__ row, const int* __restrict__ col,
                          const int* __restrict__ batch,
                          float* __restrict__ col_cnt, float* __restrict__ nb_cnt,
                          float* __restrict__ eb_cnt) {
  int i = blockIdx.x * blockDim.x + threadIdx.x;
  if (i < EE) {
    atomicAdd(&col_cnt[col[i]], 1.f);
    atomicAdd(&eb_cnt[batch[row[i]]], 1.f);
  }
  if (i < NN) atomicAdd(&nb_cnt[batch[i]], 1.f);
}

// Sum PP partial copies of a [len] buffer.
__global__ void gn_reduce_parts(const float* __restrict__ part,
                                float* __restrict__ out, int len) {
  int i = blockIdx.x * blockDim.x + threadIdx.x;
  if (i < len) {
    float s = 0.f;
    for (int p = 0; p < PP; ++p) s += part[(size_t)p * len + i];
    out[i] = s;
  }
}

__global__ void gn_add_inplace(float* __restrict__ out,
                               const float* __restrict__ a, int n) {
  int i = blockIdx.x * blockDim.x + threadIdx.x;
  if (i < n) out[i] += a[i];
}

__global__ void gn_add2(float* __restrict__ out, const float* __restrict__ a,
                        const float* __restrict__ b, int n) {
  int i = blockIdx.x * blockDim.x + threadIdx.x;
  if (i < n) out[i] = a[i] + b[i];
}

// ---------------- Host launch ------------------------------------------------
extern "C" void kernel_launch(void* const* d_in, const int* in_sizes, int n_in,
                              void* d_out, int out_size, void* d_ws, size_t ws_size,
                              hipStream_t stream) {
  const float* x0 = (const float*)d_in[0];
  const float* e0 = (const float*)d_in[1];
  const float* u0 = (const float*)d_in[2];
  const float* e_w1 = (const float*)d_in[3];
  const float* e_b1 = (const float*)d_in[4];
  const float* e_w2 = (const float*)d_in[5];
  const float* e_b2 = (const float*)d_in[6];
  const float* e_w3 = (const float*)d_in[7];
  const float* e_b3 = (const float*)d_in[8];
  const float* n1_w1 = (const float*)d_in[9];
  const float* n1_b1 = (const float*)d_in[10];
  const float* n1_w2 = (const float*)d_in[11];
  const float* n1_b2 = (const float*)d_in[12];
  const float* n2_w1 = (const float*)d_in[13];
  const float* n2_b1 = (const float*)d_in[14];
  const float* n2_w2 = (const float*)d_in[15];
  const float* n2_b2 = (const float*)d_in[16];
  const float* g_w1 = (const float*)d_in[17];
  const float* g_b1 = (const float*)d_in[18];
  const float* g_w2 = (const float*)d_in[19];
  const float* g_b2 = (const float*)d_in[20];
  const int* ei = (const int*)d_in[21];
  const int* batch = (const int*)d_in[22];
  const int* row = ei;
  const int* col = ei + EE;

  // workspace carve (256B aligned)
  char* ws = (char*)d_ws;
  size_t off = 0;
  auto carve = [&](size_t bytes) {
    char* p = ws + off;
    off = (off + bytes + 255) & ~(size_t)255;
    return p;
  };
  float* agg_sum = (float*)carve((size_t)NN * HH * 4);
  float* col_cnt = (float*)carve((size_t)NN * 4);
  float* nb_cnt = (float*)carve(BB * 4);
  float* eb_cnt = (float*)carve(BB * 4);
  float* n_agg = (float*)carve(BB * DD * 4);
  float* e_agg = (float*)carve(BB * DD * 4);
  float* e_agg_part = (float*)carve((size_t)PP * BB * DD * 4);
  float* n_agg_part = (float*)carve((size_t)PP * BB * DD * 4);
  float* ubuf0 = (float*)carve(BB * DD * 4);
  float* ubuf1 = (float*)carve(BB * DD * 4);
  __bf16* wbuf = (__bf16*)carve((size_t)LL * 163840 * 2);

  // weight bf16 conversion (transposed) — per-layer offsets inside wbuf
  for (int l = 0; l < LL; ++l) {
    __bf16* wl = wbuf + (size_t)l * 163840;
    gn_convT<<<(256 * 128 + 255) / 256, 256, 0, stream>>>(e_w1 + (size_t)l * 256 * 128, wl + 0, 256, 128);
    gn_convT<<<(128 * 128 + 255) / 256, 256, 0, stream>>>(e_w2 + (size_t)l * 128 * 128, wl + 32768, 128, 128);
    gn_convT<<<(128 * 64 + 255) / 256, 256, 0, stream>>>(e_w3 + (size_t)l * 128 * 64, wl + 49152, 128, 64);
    gn_convT<<<(128 * 128 + 255) / 256, 256, 0, stream>>>(n1_w1 + (size_t)l * 128 * 128, wl + 57344, 128, 128);
    gn_convT<<<(128 * 128 + 255) / 256, 256, 0, stream>>>(n1_w2 + (size_t)l * 128 * 128, wl + 73728, 128, 128);
    gn_convT<<<(256 * 128 + 255) / 256, 256, 0, stream>>>(n2_w1 + (size_t)l * 256 * 128, wl + 90112, 256, 128);
    gn_convT<<<(128 * 64 + 255) / 256, 256, 0, stream>>>(n2_w2 + (size_t)l * 128 * 64, wl + 122880, 128, 64);
    gn_convT<<<(192 * 128 + 255) / 256, 256, 0, stream>>>(g_w1 + (size_t)l * 192 * 128, wl + 131072, 192, 128);
    gn_convT<<<(128 * 64 + 255) / 256, 256, 0, stream>>>(g_w2 + (size_t)l * 128 * 64, wl + 155648, 128, 64);
  }

  // scatter-mean counts (recomputed every call — deterministic)
  hipMemsetAsync(col_cnt, 0, (size_t)NN * 4, stream);
  hipMemsetAsync(nb_cnt, 0, BB * 4, stream);
  hipMemsetAsync(eb_cnt, 0, BB * 4, stream);
  gn_counts<<<(EE + 255) / 256, 256, 0, stream>>>(row, col, batch, col_cnt, nb_cnt, eb_cnt);

  float* out_x = (float*)d_out;
  float* out_e = out_x + (size_t)NN * DD;
  float* out_u = out_e + (size_t)EE * DD;

  for (int l = 0; l < LL; ++l) {
    hipMemsetAsync(agg_sum, 0, (size_t)NN * HH * 4, stream);
    hipMemsetAsync(e_agg_part, 0, (size_t)PP * BB * DD * 4, stream);
    hipMemsetAsync(n_agg_part, 0, (size_t)PP * BB * DD * 4, stream);

    const float* xin = l ? out_x : x0;
    const float* ein = l ? out_e : e0;
    const float* uin = l ? ubuf0 : u0;
    float* uout = l ? ubuf1 : ubuf0;
    __bf16* wl = wbuf + (size_t)l * 163840;

    gn_edge_kernel<<<EE / 32, 32, 0, stream>>>(
        xin, ein, uin, row, col, batch,
        wl + 0, e_b1 + (size_t)l * HH, wl + 32768, e_b2 + (size_t)l * HH,
        wl + 49152, e_b3 + (size_t)l * DD,
        wl + 57344, n1_b1 + (size_t)l * HH, wl + 73728, n1_b2 + (size_t)l * HH,
        out_e, agg_sum, e_agg_part);

    gn_reduce_parts<<<(BB * DD + 255) / 256, 256, 0, stream>>>(e_agg_part, e_agg, BB * DD);

    gn_node_kernel<<<(NN + 31) / 32, 32, 0, stream>>>(
        xin, uin, agg_sum, col_cnt, batch,
        wl + 90112, n2_b1 + (size_t)l * HH, wl + 122880, n2_b2 + (size_t)l * DD,
        out_x, n_agg_part);

    gn_reduce_parts<<<(BB * DD + 255) / 256, 256, 0, stream>>>(n_agg_part, n_agg, BB * DD);

    gn_global_kernel<<<1, 32, 0, stream>>>(
        uin, n_agg, e_agg, nb_cnt, eb_cnt,
        wl + 131072, g_b1 + (size_t)l * HH, wl + 155648, g_b2 + (size_t)l * DD,
        uout);
  }

  // residuals (layer outputs are fresh assignments, so += input is replay-safe)
  gn_add_inplace<<<((size_t)NN * DD + 255) / 256, 256, 0, stream>>>(out_x, x0, NN * DD);
  gn_add_inplace<<<((size_t)EE * DD + 255) / 256, 256, 0, stream>>>(out_e, e0, EE * DD);
  gn_add2<<<(BB * DD + 255) / 256, 256, 0, stream>>>(out_u, ubuf1, u0, BB * DD);
}
